// SRAttention_56143812494020
// MI455X (gfx1250) — matrix operand order repaired and checked by reading, not compiled, not run on hardware
//
#include <hip/hip_runtime.h>
#include <hip/hip_bf16.h>

// ---------------------------------------------------------------------------
// Problem constants (from reference): B=8, Hp=Wp=128, N=16384, C=128,
// HEADS=2, HEAD_DIM=64, SR=8, Nk=256.
// ---------------------------------------------------------------------------
#define BATCH 8
#define NTOK  16384      // Hp*Wp
#define CDIM  128
#define HEADS 2
#define HDIM  64
#define NKV   256        // (128/8)^2
#define KCONV 8192       // C * SR * SR

typedef __attribute__((ext_vector_type(16))) __bf16 v16bf;
typedef __attribute__((ext_vector_type(8)))  __bf16 v8bf;
typedef __attribute__((ext_vector_type(4)))  __bf16 v4bf;
typedef __attribute__((ext_vector_type(8)))  float  v8f;

// ---------------------------------------------------------------------------
// WMMA wrapper:  D(16x16 f32) = A(16x32 bf16) * B(32x16 bf16) + C
// ---------------------------------------------------------------------------
__device__ inline v8f wmma_bf16(v16bf a, v16bf b, v8f c) {
    return __builtin_amdgcn_wmma_f32_16x16x32_bf16(
        /*neg_a=*/false, a, /*neg_b=*/false, b,
        /*c_mod=*/(short)0, c, /*reuse_a=*/false, /*reuse_b=*/false);
}

// A-fragment loader.  A0 points at row m0 of a row-major [*, lda] bf16 matrix.
// ISA layout (16-bit A 16x32): lane<16 row=lane holds K {k..k+7, k+16..k+23};
// lane>=16 same row, K {k+8..k+15, k+24..k+31}.  Both are contiguous 16B runs.
__device__ inline v16bf load_a_frag(const __bf16* __restrict__ A0, int lda, int k) {
    int lane = threadIdx.x & 31;
    const __bf16* p = A0 + (size_t)(lane & 15) * lda + k + ((lane >> 4) << 3);
    v8bf lo = *(const v8bf*)p;
    v8bf hi = *(const v8bf*)(p + 16);
    v16bf r;
#pragma unroll
    for (int e = 0; e < 8; ++e) { r[e] = lo[e]; r[e + 8] = hi[e]; }
    return r;
}

// B-fragment loader for B = W^T where W is row-major [N, ldw] (row n = column n
// of B).  ISA layout (bf16 B 32x16): lanes 0-15 hold K {k..k+7, k+16..k+23} of
// column n=lane; lanes 16-31 hold K {k+8..k+15, k+24..k+31}. Two 16B runs.
__device__ inline v16bf load_b_frag(const __bf16* __restrict__ W0, int ldw, int k) {
    int lane = threadIdx.x & 31;
    const __bf16* p = W0 + (size_t)(lane & 15) * ldw + k + ((lane >> 4) << 3);
    v8bf lo = *(const v8bf*)p;
    v8bf hi = *(const v8bf*)(p + 16);
    v16bf r;
#pragma unroll
    for (int e = 0; e < 8; ++e) { r[e] = lo[e]; r[e + 8] = hi[e]; }
    return r;
}

// C/D 16x16 f32 store: VGPR r -> (M = r + 8*(lane>=16), N = lane&15).
template <bool OUT_BF16>
__device__ inline void store_tile(void* __restrict__ out, int ldc, int m0, int n0,
                                  v8f c, const float* __restrict__ bias) {
    int lane = threadIdx.x & 31;
    int n  = lane & 15;
    int mb = (lane >> 4) << 3;
    float bi = bias ? bias[n0 + n] : 0.0f;
#pragma unroll
    for (int r = 0; r < 8; ++r) {
        float v = c[r] + bi;
        size_t idx = (size_t)(m0 + mb + r) * ldc + n0 + n;
        if (OUT_BF16) ((__bf16*)out)[idx] = (__bf16)v;
        else          ((float*)out)[idx]  = v;
    }
}

// ---------------------------------------------------------------------------
// Elementwise converters
// ---------------------------------------------------------------------------
__global__ void cvt_f32_bf16_x4(const float* __restrict__ s, __bf16* __restrict__ d, int n4) {
    int i = blockIdx.x * blockDim.x + threadIdx.x;
    if (i >= n4) return;
    const float4 f = ((const float4*)s)[i];
    v4bf o;
    o[0] = (__bf16)f.x; o[1] = (__bf16)f.y; o[2] = (__bf16)f.z; o[3] = (__bf16)f.w;
    ((v4bf*)d)[i] = o;
}

// sr_w [O=128, I=128, 8, 8] (OIHW) -> bf16 [O, K] with K = (i*8+j)*128 + c
// (patch-position-major, channel-minor — matches channel-last x gathering).
__global__ void cvt_sr_transpose(const float* __restrict__ s, __bf16* __restrict__ d) {
    int i = blockIdx.x * blockDim.x + threadIdx.x;
    if (i >= CDIM * KCONV) return;
    int o = i >> 13;          // / 8192
    int k = i & (KCONV - 1);
    int p = k >> 7;           // patch position i*8+j
    int c = k & 127;          // input channel
    d[i] = (__bf16)s[((size_t)o << 13) + (c << 6) + p];
}

// ---------------------------------------------------------------------------
// Generic GEMM:  out[M,N] = A[M,K](bf16) * W[N,K]^T(bf16) + bias
// One wave owns a 32x32 tile (2x2 WMMA accumulators, frag reuse).
// ---------------------------------------------------------------------------
template <bool OUT_BF16>
__global__ __launch_bounds__(256) void gemm_bf16(
    const __bf16* __restrict__ A, const __bf16* __restrict__ W,
    const float* __restrict__ bias, void* __restrict__ out,
    int M, int N, int K, int ldc) {
    int wid = blockIdx.x * (blockDim.x >> 5) + (threadIdx.x >> 5);
    int ntn = N >> 5;
    if (wid >= (M >> 5) * ntn) return;
    int m0 = (wid / ntn) << 5;
    int n0 = (wid % ntn) << 5;

    const __bf16* A0 = A + (size_t)m0 * K;
    const __bf16* A1 = A + (size_t)(m0 + 16) * K;
    const __bf16* W0 = W + (size_t)n0 * K;
    const __bf16* W1 = W + (size_t)(n0 + 16) * K;

    v8f c00 = {}, c01 = {}, c10 = {}, c11 = {};
    for (int k = 0; k < K; k += 32) {
        v16bf a0 = load_a_frag(A0, K, k);
        v16bf a1 = load_a_frag(A1, K, k);
        v16bf b0 = load_b_frag(W0, K, k);
        v16bf b1 = load_b_frag(W1, K, k);
        c00 = wmma_bf16(a0, b0, c00);
        c01 = wmma_bf16(a0, b1, c01);
        c10 = wmma_bf16(a1, b0, c10);
        c11 = wmma_bf16(a1, b1, c11);
    }
    store_tile<OUT_BF16>(out, ldc, m0,      n0,      c00, bias);
    store_tile<OUT_BF16>(out, ldc, m0,      n0 + 16, c01, bias);
    store_tile<OUT_BF16>(out, ldc, m0 + 16, n0,      c10, bias);
    store_tile<OUT_BF16>(out, ldc, m0 + 16, n0 + 16, c11, bias);
}

// ---------------------------------------------------------------------------
// SR conv as GEMM: tok[t=b*256+ty*16+tx, o] over K = 8192 gathered patches.
// A element (t, k): k = p*128 + c, p = i*8+j  ->  x[b, (ty*8+i)*128+(tx*8+j), c]
// A K-step of 32 never crosses a patch position, so runs stay contiguous.
// ---------------------------------------------------------------------------
__device__ inline v16bf load_a_conv(const __bf16* __restrict__ xbf, int tbase, int kblk) {
    int lane = threadIdx.x & 31;
    int t  = tbase + (lane & 15);
    int b  = t >> 8;
    int rr = t & 255;
    int ty = rr >> 4, tx = rr & 15;
    int p  = kblk >> 7;
    int pi = p >> 3, pj = p & 7;
    int c0 = (kblk & 127) + ((lane >> 4) << 3);
    int nin = (ty * 8 + pi) * 128 + (tx * 8 + pj);
    const __bf16* ptr = xbf + (((size_t)b * NTOK + nin) << 7) + c0;
    v8bf lo = *(const v8bf*)ptr;
    v8bf hi = *(const v8bf*)(ptr + 16);
    v16bf r;
#pragma unroll
    for (int e = 0; e < 8; ++e) { r[e] = lo[e]; r[e + 8] = hi[e]; }
    return r;
}

__global__ __launch_bounds__(256) void conv_sr_gemm(
    const __bf16* __restrict__ xbf, const __bf16* __restrict__ wsr,
    float* __restrict__ tok) {
    int wid = blockIdx.x * (blockDim.x >> 5) + (threadIdx.x >> 5);
    // tiles: (2048/32) x (128/32) = 64 x 4 = 256 waves
    int m0 = (wid >> 2) << 5;
    int n0 = (wid & 3)  << 5;
    const __bf16* W0 = wsr + (size_t)n0 * KCONV;
    const __bf16* W1 = wsr + (size_t)(n0 + 16) * KCONV;
    v8f c00 = {}, c01 = {}, c10 = {}, c11 = {};
    for (int k = 0; k < KCONV; k += 32) {
        v16bf a0 = load_a_conv(xbf, m0,      k);
        v16bf a1 = load_a_conv(xbf, m0 + 16, k);
        v16bf b0 = load_b_frag(W0, KCONV, k);
        v16bf b1 = load_b_frag(W1, KCONV, k);
        c00 = wmma_bf16(a0, b0, c00);
        c01 = wmma_bf16(a0, b1, c01);
        c10 = wmma_bf16(a1, b0, c10);
        c11 = wmma_bf16(a1, b1, c11);
    }
    store_tile<false>(tok, CDIM, m0,      n0,      c00, nullptr);
    store_tile<false>(tok, CDIM, m0,      n0 + 16, c01, nullptr);
    store_tile<false>(tok, CDIM, m0 + 16, n0,      c10, nullptr);
    store_tile<false>(tok, CDIM, m0 + 16, n0 + 16, c11, nullptr);
}

// ---------------------------------------------------------------------------
// LayerNorm over C=128: one wave per row (4 values per lane), shuffle reduce.
// ---------------------------------------------------------------------------
__global__ __launch_bounds__(256) void ln_bf16(
    const float* __restrict__ tok, const float* __restrict__ g,
    const float* __restrict__ beta, __bf16* __restrict__ out) {
    int row  = blockIdx.x * 8 + (threadIdx.x >> 5);
    int lane = threadIdx.x & 31;
    const float* rp = tok + (size_t)row * CDIM;
    float v[4];
#pragma unroll
    for (int e = 0; e < 4; ++e) v[e] = rp[lane + 32 * e];
    float s = v[0] + v[1] + v[2] + v[3];
#pragma unroll
    for (int off = 16; off >= 1; off >>= 1) s += __shfl_xor(s, off, 32);
    float mu = s * (1.0f / 128.0f);
    float q = 0.0f;
#pragma unroll
    for (int e = 0; e < 4; ++e) { v[e] -= mu; q += v[e] * v[e]; }
#pragma unroll
    for (int off = 16; off >= 1; off >>= 1) q += __shfl_xor(q, off, 32);
    float inv = rsqrtf(q * (1.0f / 128.0f) + 1e-5f);
    __bf16* op = out + (size_t)row * CDIM;
#pragma unroll
    for (int e = 0; e < 4; ++e) {
        int c = lane + 32 * e;
        op[c] = (__bf16)(v[e] * inv * g[c] + beta[c]);
    }
}

// ---------------------------------------------------------------------------
// Fused attention: per block = one (b,h) and 64 queries; 4 waves x 16 queries.
// K [256x64] and V^T [64x256] staged in LDS; full 16x256 score tile per wave
// in registers; half-wave shuffle softmax; P round-trips LDS to reach the
// WMMA A-fragment layout; O = P @ V. Scores never hit HBM.
// LDS: 32KB K + 32KB V^T + 32KB P = 96KB (of 320KB).
// ---------------------------------------------------------------------------
__global__ __launch_bounds__(128) void attn_fused(
    const __bf16* __restrict__ Qbf, const __bf16* __restrict__ KVbf,
    __bf16* __restrict__ Obf) {
    extern __shared__ __bf16 smem[];
    __bf16* Ks  = smem;                  // [256][64]
    __bf16* Vts = smem + NKV * HDIM;     // [64][256]
    __bf16* Ps  = smem + 2 * NKV * HDIM; // [4][16][256]

    int bh = blockIdx.x >> 8;            // 0..15
    int qb = blockIdx.x & 255;
    int b  = bh >> 1, h = bh & 1;
    int tid = threadIdx.x;

    const __bf16* kvbase = KVbf + (size_t)b * NKV * (2 * CDIM) + h * HDIM;
    // K rows -> LDS (8B per thread per iter)
    for (int idx = tid; idx < NKV * (HDIM / 4); idx += 128) {
        int j = idx >> 4, c = (idx & 15) << 2;
        *(v4bf*)(Ks + j * HDIM + c) = *(const v4bf*)(kvbase + (size_t)j * (2 * CDIM) + c);
    }
    // V transposed -> LDS
    for (int idx = tid; idx < NKV * HDIM; idx += 128) {
        int j = idx >> 6, d = idx & 63;
        Vts[d * NKV + j] = kvbase[(size_t)j * (2 * CDIM) + CDIM + d];
    }
    __syncthreads();

    int w = tid >> 5, lane = tid & 31;
    int q0 = qb * 64 + w * 16;
    const __bf16* Qp = Qbf + ((size_t)(b * NTOK + q0)) * CDIM + h * HDIM;

    // S = Q K^T : A frags reused across all 16 key tiles
    v16bf a0 = load_a_frag(Qp, CDIM, 0);
    v16bf a1 = load_a_frag(Qp, CDIM, 32);
    v8f s[16];
#pragma unroll
    for (int nt = 0; nt < 16; ++nt) {
        v8f c = {};
        c = wmma_bf16(a0, load_b_frag(Ks + nt * 16 * HDIM, HDIM, 0),  c);
        c = wmma_bf16(a1, load_b_frag(Ks + nt * 16 * HDIM, HDIM, 32), c);
        s[nt] = c;
    }

    // softmax(s * 0.125) per row; rows live in half-wave lane groups
#pragma unroll
    for (int r = 0; r < 8; ++r) {
        float mx = -3.0e38f;
#pragma unroll
        for (int nt = 0; nt < 16; ++nt) mx = fmaxf(mx, s[nt][r]);
        mx = fmaxf(mx, __shfl_xor(mx, 8, 32));
        mx = fmaxf(mx, __shfl_xor(mx, 4, 32));
        mx = fmaxf(mx, __shfl_xor(mx, 2, 32));
        mx = fmaxf(mx, __shfl_xor(mx, 1, 32));
        float sum = 0.0f;
#pragma unroll
        for (int nt = 0; nt < 16; ++nt) {
            float e = __expf((s[nt][r] - mx) * 0.125f);
            s[nt][r] = e; sum += e;
        }
        sum += __shfl_xor(sum, 8, 32);
        sum += __shfl_xor(sum, 4, 32);
        sum += __shfl_xor(sum, 2, 32);
        sum += __shfl_xor(sum, 1, 32);
        float inv = 1.0f / sum;
#pragma unroll
        for (int nt = 0; nt < 16; ++nt) s[nt][r] *= inv;
    }

    // P -> LDS (bf16) in row-major [16][256] using the C-layout mapping
    __bf16* Pw = Ps + w * 16 * NKV;
    int n = lane & 15, mb = (lane >> 4) << 3;
#pragma unroll
    for (int nt = 0; nt < 16; ++nt)
#pragma unroll
        for (int r = 0; r < 8; ++r)
            Pw[(mb + r) * NKV + nt * 16 + n] = (__bf16)s[nt][r];
    __syncthreads();

    // O = P @ V  (K-dim = 256, N = 64)
    v8f o0 = {}, o1 = {}, o2 = {}, o3 = {};
    for (int k = 0; k < NKV; k += 32) {
        v16bf ap = load_a_frag(Pw, NKV, k);
        o0 = wmma_bf16(ap, load_b_frag(Vts + 0  * NKV, NKV, k), o0);
        o1 = wmma_bf16(ap, load_b_frag(Vts + 16 * NKV, NKV, k), o1);
        o2 = wmma_bf16(ap, load_b_frag(Vts + 32 * NKV, NKV, k), o2);
        o3 = wmma_bf16(ap, load_b_frag(Vts + 48 * NKV, NKV, k), o3);
    }
    __bf16* Op = Obf + ((size_t)(b * NTOK + q0)) * CDIM + h * HDIM;
#pragma unroll
    for (int r = 0; r < 8; ++r) {
        size_t ro = (size_t)(mb + r) * CDIM;
        Op[ro + n]      = (__bf16)o0[r];
        Op[ro + 16 + n] = (__bf16)o1[r];
        Op[ro + 32 + n] = (__bf16)o2[r];
        Op[ro + 48 + n] = (__bf16)o3[r];
    }
}

// ---------------------------------------------------------------------------
// Launcher
// ---------------------------------------------------------------------------
extern "C" void kernel_launch(void* const* d_in, const int* in_sizes, int n_in,
                              void* d_out, int out_size, void* d_ws, size_t ws_size,
                              hipStream_t stream) {
    const float* x      = (const float*)d_in[0];
    // d_in[1]=Hp, d_in[2]=Wp (compile-time constants here)
    const float* q_w    = (const float*)d_in[3];
    const float* q_b    = (const float*)d_in[4];
    const float* kv_w   = (const float*)d_in[5];
    const float* kv_b   = (const float*)d_in[6];
    const float* sr_w   = (const float*)d_in[7];
    const float* ln_g   = (const float*)d_in[8];
    const float* ln_b   = (const float*)d_in[9];
    const float* proj_w = (const float*)d_in[10];
    const float* proj_b = (const float*)d_in[11];

    const size_t NXC = (size_t)BATCH * NTOK * CDIM;   // 16,777,216
    const size_t NKC = (size_t)BATCH * NKV;           // 2048 rows

    char* p = (char*)d_ws;
    __bf16* xbf   = (__bf16*)p;  p += NXC * 2;                 // x in bf16
    __bf16* qbf   = (__bf16*)p;  p += NXC * 2;                 // Q
    __bf16* obf   = (__bf16*)p;  p += NXC * 2;                 // attn out
    __bf16* wqb   = (__bf16*)p;  p += (size_t)CDIM * CDIM * 2;
    __bf16* wkvb  = (__bf16*)p;  p += (size_t)2 * CDIM * CDIM * 2;
    __bf16* wprb  = (__bf16*)p;  p += (size_t)CDIM * CDIM * 2;
    __bf16* wsrb  = (__bf16*)p;  p += (size_t)CDIM * KCONV * 2;
    float*  tok   = (float*)p;   p += NKC * CDIM * 4;          // conv out (pre-LN)
    __bf16* tokbf = (__bf16*)p;  p += NKC * CDIM * 2;          // post-LN
    __bf16* kvbf  = (__bf16*)p;  p += NKC * 2 * CDIM * 2;      // K|V

    // 1) bf16 conversions
    cvt_f32_bf16_x4<<<(NXC / 4 + 255) / 256, 256, 0, stream>>>(x, xbf, (int)(NXC / 4));
    cvt_f32_bf16_x4<<<16, 256, 0, stream>>>(q_w,    wqb,  CDIM * CDIM / 4);
    cvt_f32_bf16_x4<<<32, 256, 0, stream>>>(kv_w,   wkvb, 2 * CDIM * CDIM / 4);
    cvt_f32_bf16_x4<<<16, 256, 0, stream>>>(proj_w, wprb, CDIM * CDIM / 4);
    cvt_sr_transpose<<<(CDIM * KCONV + 255) / 256, 256, 0, stream>>>(sr_w, wsrb);

    // 2) Q = x @ q_w^T + q_b   [131072 x 128] (bf16 out)
    gemm_bf16<true><<<2048, 256, 0, stream>>>(xbf, wqb, q_b, qbf,
                                              BATCH * NTOK, CDIM, CDIM, CDIM);

    // 3) SR conv as GEMM -> tok f32 [2048 x 128]
    conv_sr_gemm<<<32, 256, 0, stream>>>(xbf, wsrb, tok);

    // 4) LayerNorm -> tokbf
    ln_bf16<<<(int)(NKC / 8), 256, 0, stream>>>(tok, ln_g, ln_b, tokbf);

    // 5) KV = tokbf @ kv_w^T + kv_b  [2048 x 256] (bf16 out)
    gemm_bf16<true><<<64, 256, 0, stream>>>(tokbf, wkvb, kv_b, kvbf,
                                            (int)NKC, 2 * CDIM, CDIM, 2 * CDIM);

    // 6) fused attention (B*H*256 blocks, 96KB dynamic LDS)
    attn_fused<<<BATCH * HEADS * 256, 128, 3 * NKV * HDIM * 2, stream>>>(qbf, kvbf, obf);

    // 7) out = O @ proj_w^T + proj_b (f32 out)
    gemm_bf16<false><<<2048, 256, 0, stream>>>(obf, wprb, proj_b, d_out,
                                               BATCH * NTOK, CDIM, CDIM, CDIM);
}
